// HippocampalMoE_64974265253912
// MI455X (gfx1250) — matrix operand order repaired and checked by reading, not compiled, not run on hardware
//
#include <hip/hip_runtime.h>
#include <hip/hip_bf16.h>
#include <math.h>

typedef __attribute__((ext_vector_type(16))) _Float16 v16h;
typedef __attribute__((ext_vector_type(4)))  _Float16 v4h;
typedef __attribute__((ext_vector_type(8)))  float    v8f;

#define S_TILE 64
#define LDA 40      // staged K-slice row stride (32 + 8 pad, f16 elems)
#define LDH 136     // h1 row stride (128 + 8 pad, f16 elems)

// Shared-memory union (bytes):
//   staging: sA 2 bufs * 5120B  @ [0, 10240)
//            sB 2 bufs * 20480B @ [10240, 51200)
//   h1     : 2 * 64 * 136 * 2B = 34816B @ [0, 34816)  (aliased after K-loop)
#define SMEM_BYTES 51200

// ---- WMMA helpers (CDNA5 v_wmma_f32_16x16x32_f16, wave32) ----

__device__ __forceinline__ v8f wmma_f16(v16h a, v16h b, v8f c) {
  return __builtin_amdgcn_wmma_f32_16x16x32_f16(
      false, a, false, b, (short)0, c, false, false);
}

// A-matrix 16x32 f16 fragment (this code's contiguous K order, same as B): lane L holds row M=L&15;
// half = L>>4 selects K in {0..15} (half 0) or {16..31} (half 1); element i holds K = i + 16*half.
__device__ __forceinline__ v16h load_a_frag(const _Float16* base, int ld) {
  const int lane = threadIdx.x & 31;
  const _Float16* row = base + (lane & 15) * ld + ((lane >> 4) << 4);
  v16h a;
#pragma unroll
  for (int i = 0; i < 8; ++i) a[i] = row[i];
#pragma unroll
  for (int i = 0; i < 8; ++i) a[8 + i] = row[8 + i];
  return a;
}

// B-matrix 32x16 f16 fragment: lane L holds column N=L&15; element i holds
// K = i + 16*(L>>4). Source stored as [N][K] row-major with stride ld.
__device__ __forceinline__ v16h load_b_frag_lds(const _Float16* base, int ld) {
  const int lane = threadIdx.x & 31;
  const _Float16* p = base + (lane & 15) * ld + ((lane >> 4) << 4);
  v16h b;
#pragma unroll
  for (int i = 0; i < 16; ++i) b[i] = p[i];
  return b;
}

// B fragment from global f32 [N][K] row-major, converting to f16.
__device__ __forceinline__ v16h load_b_frag_g32(const float* p) {
  const float4* q = (const float4*)p;
  float4 a0 = q[0], a1 = q[1], a2 = q[2], a3 = q[3];
  v16h b;
  b[0]=(_Float16)a0.x; b[1]=(_Float16)a0.y; b[2]=(_Float16)a0.z; b[3]=(_Float16)a0.w;
  b[4]=(_Float16)a1.x; b[5]=(_Float16)a1.y; b[6]=(_Float16)a1.z; b[7]=(_Float16)a1.w;
  b[8]=(_Float16)a2.x; b[9]=(_Float16)a2.y; b[10]=(_Float16)a2.z; b[11]=(_Float16)a2.w;
  b[12]=(_Float16)a3.x; b[13]=(_Float16)a3.y; b[14]=(_Float16)a3.z; b[15]=(_Float16)a3.w;
  return b;
}

__device__ __forceinline__ v4h cvt4(float4 v) {
  v4h h;
  h[0] = (_Float16)v.x; h[1] = (_Float16)v.y;
  h[2] = (_Float16)v.z; h[3] = (_Float16)v.w;
  return h;
}

// Stage one 32-wide K-slice: hidden[64][32] + W1[e0|e1][128][32] -> registers.
__device__ __forceinline__ void stage_load(const float* __restrict__ hsrc,
                                           const float* __restrict__ w1e0,
                                           const float* __restrict__ w1e1,
                                           int kk, int t,
                                           float4* rA, float4* rB) {
#pragma unroll
  for (int i = 0; i < 2; ++i) {               // 512 float4 / 256 thr
    int id = t + 256 * i; int r = id >> 3, c = (id & 7) << 2;
    rA[i] = *(const float4*)(hsrc + (size_t)r * 2048 + kk + c);
  }
#pragma unroll
  for (int i = 0; i < 4; ++i) {               // 1024 float4 / 256 thr
    int id = t + 256 * i; int f = id >> 3, c = (id & 7) << 2;
    rB[i] = *(const float4*)(w1e0 + (size_t)f * 2048 + kk + c);
  }
#pragma unroll
  for (int i = 0; i < 4; ++i) {
    int id = t + 256 * i; int f = id >> 3, c = (id & 7) << 2;
    rB[4 + i] = *(const float4*)(w1e1 + (size_t)f * 2048 + kk + c);
  }
}

// Registers -> f16 LDS, 8-byte ds_store_b64 packed stores.
__device__ __forceinline__ void stage_store(_Float16* sAb, _Float16* sBb,
                                            int t, const float4* rA,
                                            const float4* rB) {
#pragma unroll
  for (int i = 0; i < 2; ++i) {
    int id = t + 256 * i; int r = id >> 3, c = (id & 7) << 2;
    *(v4h*)(sAb + r * LDA + c) = cvt4(rA[i]);
  }
#pragma unroll
  for (int e = 0; e < 2; ++e)
#pragma unroll
    for (int i = 0; i < 4; ++i) {
      int id = t + 256 * i; int f = id >> 3, c = (id & 7) << 2;
      *(v4h*)(sBb + e * 128 * LDA + f * LDA + c) = cvt4(rB[e * 4 + i]);
    }
}

// ---------------- kernel 1: partial mean-pool over S ----------------
__global__ __launch_bounds__(256) void moe_pool_partial(
    const float* __restrict__ hidden, float* __restrict__ partial) {
  const int id = blockIdx.x;
  const int b  = id >> 6;
  const int hc = (id & 63) >> 3;
  const int sc = id & 7;
  const int h  = hc * 256 + threadIdx.x;
  const float* p = hidden + ((size_t)b * 4096 + (size_t)sc * 512) * 2048 + h;
  float s = 0.0f;
  for (int i = 0; i < 512; ++i) s += p[(size_t)i * 2048];
  partial[(size_t)sc * 16384 + (size_t)b * 2048 + h] = s;
}

// ---------------- kernel 2: reduce partials -> pooled sums ----------------
__global__ __launch_bounds__(256) void moe_pool_reduce(
    const float* __restrict__ partial, float* __restrict__ pooled) {
  const int idx = blockIdx.x * 256 + threadIdx.x;
  float s = 0.0f;
#pragma unroll
  for (int c = 0; c < 8; ++c) s += partial[(size_t)c * 16384 + idx];
  pooled[idx] = s;
}

// ---------------- kernel 3: router (1 block; wave b = batch row b) --------
__global__ __launch_bounds__(256) void moe_router(
    const float* __restrict__ pooled, const float* __restrict__ rw,
    const float* __restrict__ rb, const float* __restrict__ counts,
    const float* __restrict__ drift, int* __restrict__ gidx,
    float* __restrict__ gprob, float* __restrict__ tail) {
  const int t = threadIdx.x;
  const int b = t >> 5, e = t & 31;

  const float* pb = pooled + (size_t)b * 2048;
  const float* we = rw + (size_t)e * 2048;
  float dot = 0.0f;
  for (int h = 0; h < 2048; ++h) dot += pb[h] * we[h];
  float logit = dot * (1.0f / 4096.0f) + rb[e];

  float total = 0.0f, cmax = 0.0f, dmax = 0.0f;
  for (int i = 0; i < 32; ++i) {
    float c = counts[i];
    total += c;
    cmax = fmaxf(cmax, c);
    dmax = fmaxf(dmax, drift[i]);
  }
  float usage = counts[e] / fmaxf(total, 1e-8f);
  float bonus = (total > 0.0f) ? 0.1f * (1.0f - usage) : 0.0f;
  float un = counts[e] / fmaxf(cmax, 1e-8f);
  float dn = drift[e] / fmaxf(dmax, 1e-8f);
  logit += bonus - 0.05f * (un + dn);

  float m = logit;
#pragma unroll
  for (int off = 16; off > 0; off >>= 1) m = fmaxf(m, __shfl_xor(m, off));
  float ex = expf(logit - m);
  float s = ex;
#pragma unroll
  for (int off = 16; off > 0; off >>= 1) s += __shfl_xor(s, off);
  float prob = ex / s;
  tail[32 + b * 32 + e] = prob;              // router_probs [B,E]

  float v1 = prob; int i1 = e;
#pragma unroll
  for (int off = 16; off > 0; off >>= 1) {
    float ov = __shfl_xor(v1, off); int oi = __shfl_xor(i1, off);
    if (ov > v1 || (ov == v1 && oi < i1)) { v1 = ov; i1 = oi; }
  }
  float v2 = (e == i1) ? -1.0f : prob; int i2 = e;
#pragma unroll
  for (int off = 16; off > 0; off >>= 1) {
    float ov = __shfl_xor(v2, off); int oi = __shfl_xor(i2, off);
    if (ov > v2 || (ov == v2 && oi < i2)) { v2 = ov; i2 = oi; }
  }
  if (e == 0) {
    float denom = fmaxf(v1 + v2, 1e-8f);
    float q0 = v1 / denom, q1 = v2 / denom;
    gidx[2 * b] = i1; gidx[2 * b + 1] = i2;
    gprob[2 * b] = q0; gprob[2 * b + 1] = q1;
    tail[2 * b] = (float)i1;                 // topk_idx  [B,2]
    tail[2 * b + 1] = (float)i2;
    tail[16 + 2 * b] = q0;                   // topk_probs [B,2]
    tail[16 + 2 * b + 1] = q1;
  }
}

// ---------------- kernel 4: fused dual-expert FFN ----------------
// grid = B * (S/64) = 512 blocks, 256 threads (8 waves).
__global__ __launch_bounds__(256) void moe_ffn(
    const float* __restrict__ hidden, const float* __restrict__ w1,
    const float* __restrict__ b1, const float* __restrict__ w2,
    const float* __restrict__ b2, const int* __restrict__ gidx,
    const float* __restrict__ gprob, float* __restrict__ out) {
  __shared__ __align__(16) unsigned char smem[SMEM_BYTES];
  _Float16* sAr = (_Float16*)smem;             // + buf*2560 (elems)
  _Float16* sBr = (_Float16*)(smem + 10240);   // + buf*10240 (elems)
  _Float16* sH  = (_Float16*)smem;             // aliased after K-loop

  const int t = threadIdx.x;
  const int w = t >> 5;
  const int lane = t & 31;
  const int b = blockIdx.x >> 6;
  const int s0 = (blockIdx.x & 63) * S_TILE;

  const int e0 = gidx[2 * b], e1 = gidx[2 * b + 1];
  const float p0 = gprob[2 * b], p1 = gprob[2 * b + 1];
  const int   eidx[2] = {e0, e1};
  const float pw[2]   = {p0, p1};

  const size_t H = 2048, F = 128, S = 4096;
  const float* hsrc = hidden + ((size_t)b * S + s0) * H;
  const float* w1e0 = w1 + (size_t)e0 * F * H;
  const float* w1e1 = w1 + (size_t)e1 * F * H;

  const int m0 = (w & 3) * 16;
  const int nh = w >> 2;

  // ---------------- GEMM1 (double-buffered pipeline) ----------------
  v8f acc[2][4];
#pragma unroll
  for (int e = 0; e < 2; ++e)
#pragma unroll
    for (int n = 0; n < 4; ++n)
#pragma unroll
      for (int j = 0; j < 8; ++j) acc[e][n][j] = 0.0f;

  float4 rA[2], rB[8];
  stage_load(hsrc, w1e0, w1e1, 0, t, rA, rB);
  stage_store(sAr, sBr, t, rA, rB);

  for (int step = 0; step < 64; ++step) {
    if (step + 1 < 64)                               // overlap: next K-slice
      stage_load(hsrc, w1e0, w1e1, (step + 1) * 32, t, rA, rB);
    __syncthreads();                                 // buf[step&1] visible
    const _Float16* sAb = sAr + (step & 1) * (S_TILE * LDA);
    const _Float16* sBb = sBr + (step & 1) * (2 * 128 * LDA);

    v16h a = load_a_frag(sAb + m0 * LDA, LDA);
    v16h bf[8];
#pragma unroll
    for (int e = 0; e < 2; ++e)
#pragma unroll
      for (int nt = 0; nt < 4; ++nt)
        bf[e * 4 + nt] =
            load_b_frag_lds(sBb + (e * 128 + nh * 64 + nt * 16) * LDA, LDA);
#pragma unroll
    for (int e = 0; e < 2; ++e)
#pragma unroll
      for (int nt = 0; nt < 4; ++nt)
        acc[e][nt] = wmma_f16(a, bf[e * 4 + nt], acc[e][nt]);

    if (step + 1 < 64)
      stage_store(sAr + ((step + 1) & 1) * (S_TILE * LDA),
                  sBr + ((step + 1) & 1) * (2 * 128 * LDA), t, rA, rB);
  }
  __syncthreads();   // staging dead; safe to alias with sH

  // epilogue: +b1, exact-erf GELU, fold gate prob, store f16 h1 to LDS
#pragma unroll
  for (int e = 0; e < 2; ++e) {
    const float* b1e = b1 + (size_t)eidx[e] * F;
    _Float16* hdst = sH + e * S_TILE * LDH;
#pragma unroll
    for (int nt = 0; nt < 4; ++nt) {
      int f = nh * 64 + nt * 16 + (lane & 15);
      float bias = b1e[f];
#pragma unroll
      for (int j = 0; j < 8; ++j) {
        int m = m0 + j + ((lane >> 4) << 3);
        float x = acc[e][nt][j] + bias;
        float g = 0.5f * x * (1.0f + erff(x * 0.70710678118654752f));
        hdst[m * LDH + f] = (_Float16)(g * pw[e]);
      }
    }
  }
  __syncthreads();

  // ---------------- GEMM2: delta = sum_e (p_e*h1_e) @ W2[e]^T + bias ------
  v16h aF[2][4];
#pragma unroll
  for (int e = 0; e < 2; ++e)
#pragma unroll
    for (int ks = 0; ks < 4; ++ks)
      aF[e][ks] = load_a_frag(sH + e * S_TILE * LDH + m0 * LDH + ks * 32, LDH);

  const float* w2e0 = w2 + (size_t)e0 * H * F;
  const float* w2e1 = w2 + (size_t)e1 * H * F;
  float* outb = out + ((size_t)b * S + s0) * H;

  for (int tt = 0; tt < 64; ++tt) {
    const int n0 = (nh + 2 * tt) * 16;
    const int hcol = n0 + (lane & 15);
    const float* pb0 = w2e0 + (size_t)hcol * F + ((lane >> 4) << 4);
    const float* pb1 = w2e1 + (size_t)hcol * F + ((lane >> 4) << 4);

    if (tt + 1 < 64) {                       // keep W2 stream ahead in L2/L0
      const int hn = (nh + 2 * (tt + 1)) * 16 + (lane & 15);
      __builtin_prefetch(w2e0 + (size_t)hn * F, 0, 1);
      __builtin_prefetch(w2e1 + (size_t)hn * F, 0, 1);
    }

    v16h bg[8];
#pragma unroll
    for (int ks = 0; ks < 4; ++ks) bg[ks]     = load_b_frag_g32(pb0 + ks * 32);
#pragma unroll
    for (int ks = 0; ks < 4; ++ks) bg[4 + ks] = load_b_frag_g32(pb1 + ks * 32);

    v8f c;
#pragma unroll
    for (int j = 0; j < 8; ++j) c[j] = 0.0f;
#pragma unroll
    for (int ks = 0; ks < 4; ++ks) c = wmma_f16(aF[0][ks], bg[ks], c);
#pragma unroll
    for (int ks = 0; ks < 4; ++ks) c = wmma_f16(aF[1][ks], bg[4 + ks], c);

    const float bb = p0 * b2[(size_t)e0 * H + hcol] +
                     p1 * b2[(size_t)e1 * H + hcol];
#pragma unroll
    for (int j = 0; j < 8; ++j) {
      int m = m0 + j + ((lane >> 4) << 3);
      outb[(size_t)m * H + hcol] = c[j] + bb;
    }
  }
}

extern "C" void kernel_launch(void* const* d_in, const int* in_sizes, int n_in,
                              void* d_out, int out_size, void* d_ws, size_t ws_size,
                              hipStream_t stream) {
  (void)in_sizes; (void)n_in; (void)out_size; (void)ws_size;
  const float* hidden = (const float*)d_in[0];
  const float* rw     = (const float*)d_in[1];
  const float* rb     = (const float*)d_in[2];
  const float* w1     = (const float*)d_in[3];
  const float* b1     = (const float*)d_in[4];
  const float* w2     = (const float*)d_in[5];
  const float* b2     = (const float*)d_in[6];
  const float* counts = (const float*)d_in[7];
  const float* drift  = (const float*)d_in[8];
  float* out = (float*)d_out;

  float* ws_f    = (float*)d_ws;
  float* pooled  = ws_f;                    // 16384 floats (B*H sums)
  int*   gidx    = (int*)(ws_f + 16384);    // 16 ints
  float* gprob   = ws_f + 16384 + 16;       // 16 floats
  float* partial = ws_f + 16384 + 32;       // 8*16384 floats

  const unsigned long long N_DELTA = 8ULL * 4096ULL * 2048ULL;  // 67108864

  moe_pool_partial<<<512, 256, 0, stream>>>(hidden, partial);
  moe_pool_reduce<<<64, 256, 0, stream>>>(partial, pooled);
  moe_router<<<1, 256, 0, stream>>>(pooled, rw, rb, counts, drift,
                                    gidx, gprob, out + N_DELTA);
  moe_ffn<<<512, 256, 0, stream>>>(hidden, w1, b1, w2, b2, gidx, gprob, out);
}
